// SwinTransformerBlock_4621384810903
// MI455X (gfx1250) — compile-verified, matrix operand order checked by reading, not run on hardware
//
#include <hip/hip_runtime.h>
#include <hip/hip_bf16.h>
#include <math.h>

// ---------------------------------------------------------------------------
// SwinV2 block for MI455X (gfx1250, wave32, WMMA bf16 + async-to-LDS path)
// ---------------------------------------------------------------------------

typedef __bf16 bf16_t;
typedef __bf16 v16bf __attribute__((ext_vector_type(16)));
typedef float  v8f   __attribute__((ext_vector_type(8)));

union Frag { unsigned u[8]; v16bf v; };

#define TOK 100352   // B*H*W = 32*56*56
#define CDIM 384
#define NHEADS 12
#define HD 32
#define NWIN 2048    // B * 64 windows
#define WTOK 49

#if defined(__has_builtin)
#if __has_builtin(__builtin_amdgcn_global_load_async_to_lds_b128)
#define USE_ASYNC_LDS 1
#endif
#if __has_builtin(__builtin_amdgcn_s_wait_asynccnt)
#define HAVE_WAIT_ASYNC 1
#endif
#endif

#if defined(USE_ASYNC_LDS)
// builtin prototype (from hipcc diagnostic): (v4i __device__*, v4i __shared__*, Ii, Ii)
typedef int v4i_async __attribute__((vector_size(16)));
typedef __attribute__((address_space(1))) v4i_async* async_gptr;
typedef __attribute__((address_space(3))) v4i_async* async_lptr;
#endif

__device__ __forceinline__ void wait_async0() {
#if defined(HAVE_WAIT_ASYNC)
  __builtin_amdgcn_s_wait_asynccnt(0);
#elif defined(USE_ASYNC_LDS)
  asm volatile("s_wait_asynccnt 0x0" ::: "memory");
#endif
}

// Load a 16x32 bf16 fragment (A layout, or Bt[n][k] layout for B) from LDS.
// Per ISA 7.12.2:
//   lanes 0-15: M=lane, VGPR r holds K = (r>=4?16:0) + 2*(r&3) + {0,1}
//   lanes16-31: same M, K += 8
__device__ __forceinline__ void load_frag(const bf16_t* base, int stride,
                                          int lane, Frag& f) {
  const int m   = lane & 15;
  const int khi = (lane & 16) ? 8 : 0;
  const bf16_t* row = base + m * stride;
#pragma unroll
  for (int r = 0; r < 8; ++r) {
    const int k = ((r & 4) ? 16 : 0) + khi + ((r & 3) << 1);
    f.u[r] = *(const unsigned*)(row + k);
  }
}

// ---------------------------------------------------------------------------
// fp32 -> bf16 weight conversion
// ---------------------------------------------------------------------------
__global__ void f2bf_kernel(const float* __restrict__ in, bf16_t* __restrict__ out, int n) {
  for (int i = blockIdx.x * blockDim.x + threadIdx.x; i < n; i += gridDim.x * blockDim.x)
    out[i] = (bf16_t)in[i];
}

// ---------------------------------------------------------------------------
// CPB: tab[p][head] = relu(rel(p) @ w1^T + b1) @ w2^T     (169 x 12)
// ---------------------------------------------------------------------------
__global__ __launch_bounds__(128)
void cpb1_kernel(const float* __restrict__ w1, const float* __restrict__ b1,
                 const float* __restrict__ w2, float* __restrict__ tab) {
  __shared__ float red[128][12];
  const int p = blockIdx.x;            // 0..168
  const int i = p / 13, j = p % 13;
  const float c0 = (i - 6) / 6.0f, c1 = (j - 6) / 6.0f;
  const float v0 = 8.f * c0, v1 = 8.f * c1;
  const float r0 = copysignf(log2f(fabsf(v0) + 1.f) / 3.f, v0);
  const float r1 = copysignf(log2f(fabsf(v1) + 1.f) / 3.f, v1);
  float part[12];
#pragma unroll
  for (int h = 0; h < 12; ++h) part[h] = 0.f;
  for (int jj = threadIdx.x; jj < 512; jj += 128) {
    const float hv = fmaxf(0.f, w1[jj * 2] * r0 + w1[jj * 2 + 1] * r1 + b1[jj]);
#pragma unroll
    for (int h = 0; h < 12; ++h) part[h] += hv * w2[h * 512 + jj];
  }
#pragma unroll
  for (int h = 0; h < 12; ++h) red[threadIdx.x][h] = part[h];
  __syncthreads();
  for (int s = 64; s > 0; s >>= 1) {
    if (threadIdx.x < (unsigned)s)
#pragma unroll
      for (int h = 0; h < 12; ++h) red[threadIdx.x][h] += red[threadIdx.x + s][h];
    __syncthreads();
  }
  if (threadIdx.x < 12) tab[p * 12 + threadIdx.x] = red[0][threadIdx.x];
}

// rpb[h][m][n] = 16*sigmoid(tab[idx(m,n)][h])
__global__ void cpb2_kernel(const float* __restrict__ tab, float* __restrict__ rpb) {
  const int idx = blockIdx.x * blockDim.x + threadIdx.x;
  if (idx >= 12 * 49 * 49) return;
  const int h = idx / 2401;
  const int r = idx % 2401;
  const int m = r / 49, n = r % 49;
  const int dr = m / 7 - n / 7 + 6;
  const int dc = m % 7 - n % 7 + 6;
  const float t = tab[(dr * 13 + dc) * 12 + h];
  rpb[idx] = 16.f / (1.f + __expf(-t));
}

// ---------------------------------------------------------------------------
// LayerNorm (+ optional shift/window-partition) -> bf16 rows (TOK, 384)
// ---------------------------------------------------------------------------
template<bool WINDOWED>
__global__ __launch_bounds__(128)
void ln_kernel(const float* __restrict__ x, const float* __restrict__ g,
               const float* __restrict__ bta, bf16_t* __restrict__ out) {
  __shared__ float s1[128], s2[128];
  const int m = blockIdx.x;
  const int tid = threadIdx.x;
  size_t src;
  if (WINDOWED) {
    const int b = m / 3136;           // 3136 = 64*49
    const int rem = m % 3136;
    const int wnd = rem / 49, n = rem % 49;
    const int wh = wnd >> 3, ww = wnd & 7;
    const int i = n / 7, j = n % 7;
    const int hh = (wh * 7 + i + 3) % 56;
    const int wc = (ww * 7 + j + 3) % 56;
    src = ((size_t)(b * 56 + hh) * 56 + wc) * CDIM;
  } else {
    src = (size_t)m * CDIM;
  }
  float v[3], a = 0.f, bb = 0.f;
#pragma unroll
  for (int i = 0; i < 3; ++i) {
    v[i] = x[src + tid + i * 128];
    a += v[i];
    bb += v[i] * v[i];
  }
  s1[tid] = a; s2[tid] = bb;
  __syncthreads();
  for (int s = 64; s > 0; s >>= 1) {
    if (tid < s) { s1[tid] += s1[tid + s]; s2[tid] += s2[tid + s]; }
    __syncthreads();
  }
  const float mu = s1[0] * (1.f / CDIM);
  const float var = s2[0] * (1.f / CDIM) - mu * mu;
  const float rstd = rsqrtf(var + 1e-5f);
#pragma unroll
  for (int i = 0; i < 3; ++i) {
    const int c = tid + i * 128;
    out[(size_t)m * CDIM + c] = (bf16_t)((v[i] - mu) * rstd * g[c] + bta[c]);
  }
}

// ---------------------------------------------------------------------------
// Tiled WMMA GEMM: C(M,N) = A(M,K) @ Bt(N,K)^T (+bias, epilogue by MODE)
//   MODE 0: bias            -> bf16 out
//   MODE 1: bias + GELU     -> bf16 out
//   MODE 2: bias            -> fp32 out
//   MODE 3: bias + residual -> fp32 out
// block = 256 threads (8 waves), BM=BN=128, BK=32, double-buffered LDS.
// Tile staging uses GLOBAL_LOAD_ASYNC_TO_LDS_B128 (ASYNCcnt) when available.
// ---------------------------------------------------------------------------
template<int MODE>
__global__ __launch_bounds__(256)
void gemm_bf16_wmma(const bf16_t* __restrict__ A, const bf16_t* __restrict__ Bt,
                    const float* __restrict__ bias, const float* __restrict__ resid,
                    void* __restrict__ outp, int M, int N, int K) {
  constexpr int BM = 128, BN = 128, BK = 32, LSTR = 40; // 80B stride: aligned + bank-spread
  __shared__ __align__(16) bf16_t As[2][BM * LSTR];
  __shared__ __align__(16) bf16_t Bs[2][BN * LSTR];

  const int tid  = threadIdx.x;
  const int lane = tid & 31;
  const int wave = tid >> 5;
  const int mblk = blockIdx.y * BM;
  const int nblk = blockIdx.x * BN;
  const int wm = (wave >> 1) * 32;   // 4 wave-rows
  const int wn = (wave & 1) * 64;    // 2 wave-cols

  const v8f vzero = {0.f, 0.f, 0.f, 0.f, 0.f, 0.f, 0.f, 0.f};
  v8f acc[2][4];
#pragma unroll
  for (int mt = 0; mt < 2; ++mt)
#pragma unroll
    for (int nt = 0; nt < 4; ++nt) acc[mt][nt] = vzero;

  const int nk = K / BK;

#if defined(USE_ASYNC_LDS)
  auto issue_tile = [&](int kt, int buf) {
#pragma unroll
    for (int c = 0; c < 2; ++c) {
      const int ch = tid + c * 256;       // 512 16B-chunks per tile
      const int row = ch >> 2;
      const int col = (ch & 3) * 8;
      __builtin_amdgcn_global_load_async_to_lds_b128(
          (async_gptr)(A + (size_t)(mblk + row) * K + kt * BK + col),
          (async_lptr)&As[buf][row * LSTR + col], 0, 0);
      __builtin_amdgcn_global_load_async_to_lds_b128(
          (async_gptr)(Bt + (size_t)(nblk + row) * K + kt * BK + col),
          (async_lptr)&Bs[buf][row * LSTR + col], 0, 0);
    }
  };
  issue_tile(0, 0);
#else
  uint4 ra[2], rb[2];
  auto g_load = [&](int kt) {
#pragma unroll
    for (int c = 0; c < 2; ++c) {
      const int ch = tid + c * 256;
      const int row = ch >> 2;
      const int col = (ch & 3) * 8;
      ra[c] = *(const uint4*)(A + (size_t)(mblk + row) * K + kt * BK + col);
      rb[c] = *(const uint4*)(Bt + (size_t)(nblk + row) * K + kt * BK + col);
    }
  };
  auto s_store = [&](int buf) {
#pragma unroll
    for (int c = 0; c < 2; ++c) {
      const int ch = tid + c * 256;
      const int row = ch >> 2;
      const int col = (ch & 3) * 8;
      *(uint4*)(&As[buf][row * LSTR + col]) = ra[c];
      *(uint4*)(&Bs[buf][row * LSTR + col]) = rb[c];
    }
  };
  g_load(0);
  s_store(0);
#endif

  int buf = 0;
  for (int kt = 0; kt < nk; ++kt) {
#if defined(USE_ASYNC_LDS)
    wait_async0();          // tile kt resident in LDS (this wave's copies)
    __syncthreads();        // all waves' copies visible; prior compute on buf^1 done
    if (kt + 1 < nk) issue_tile(kt + 1, buf ^ 1);
#else
    __syncthreads();
    if (kt + 1 < nk) g_load(kt + 1);
#endif
    if (kt + 2 < nk) {  // CDNA5 global_prefetch_b8 for the tile after next
      __builtin_prefetch(A + (size_t)(mblk + (tid >> 1)) * K + (kt + 2) * BK, 0, 1);
      __builtin_prefetch(Bt + (size_t)(nblk + (tid >> 1)) * K + (kt + 2) * BK, 0, 1);
    }
    Frag af[2], bfr[4];
#pragma unroll
    for (int mt = 0; mt < 2; ++mt)
      load_frag(&As[buf][(wm + mt * 16) * LSTR], LSTR, lane, af[mt]);
#pragma unroll
    for (int nt = 0; nt < 4; ++nt)
      load_frag(&Bs[buf][(wn + nt * 16) * LSTR], LSTR, lane, bfr[nt]);
#pragma unroll
    for (int mt = 0; mt < 2; ++mt)
#pragma unroll
      for (int nt = 0; nt < 4; ++nt)
        acc[mt][nt] = __builtin_amdgcn_wmma_f32_16x16x32_bf16(
            false, af[mt].v, false, bfr[nt].v, (short)0, acc[mt][nt], false, false);
#if !defined(USE_ASYNC_LDS)
    if (kt + 1 < nk) {
      __syncthreads();
      s_store(buf ^ 1);
    }
#endif
    buf ^= 1;
  }

  // epilogue: D lane layout => element e: row = e + (lane>=16 ? 8 : 0), col = lane&15
#pragma unroll
  for (int mt = 0; mt < 2; ++mt) {
#pragma unroll
    for (int nt = 0; nt < 4; ++nt) {
      const int n = nblk + wn + nt * 16 + (lane & 15);
      const int mbase = mblk + wm + mt * 16 + ((lane & 16) ? 8 : 0);
      const float bv = bias[n];
#pragma unroll
      for (int e = 0; e < 8; ++e) {
        const int m = mbase + e;
        const size_t idx = (size_t)m * N + n;
        float v = acc[mt][nt][e] + bv;
        if (MODE == 0) {
          ((bf16_t*)outp)[idx] = (bf16_t)v;
        } else if (MODE == 1) {
          v = 0.5f * v * (1.f + erff(v * 0.70710678118654752f));
          ((bf16_t*)outp)[idx] = (bf16_t)v;
        } else if (MODE == 2) {
          ((float*)outp)[idx] = v;
        } else {
          ((float*)outp)[idx] = v + resid[idx];
        }
      }
    }
  }
}

// ---------------------------------------------------------------------------
// Windowed cosine attention, one (window, head) per block (128 threads, 4 waves)
// N=49 padded to 64; QK^T: 4x4 WMMA tiles (K=32); AV: 4x2 tiles (K=64).
// rpb+mask pre-staged into LDS so the WMMA epilogue is branch-free.
// ---------------------------------------------------------------------------
__global__ __launch_bounds__(128)
void attn_kernel(const bf16_t* __restrict__ qkv, const float* __restrict__ rpb,
                 const float* __restrict__ mask, const float* __restrict__ lscale,
                 bf16_t* __restrict__ o) {
  constexpr int LQ = 40;   // qn/kn row stride (halves)
  constexpr int LV = 72;   // vt row stride (tokens)
  constexpr int LT = 72;   // attn row stride
  __shared__ __align__(16) bf16_t qn[64 * LQ];
  __shared__ __align__(16) bf16_t kn[64 * LQ];
  __shared__ __align__(16) bf16_t vt[32 * LV];   // transposed v: [dim][token]
  __shared__ __align__(16) bf16_t at[64 * LT];
  __shared__ float lg[64 * 65];

  const int tid  = threadIdx.x;
  const int lane = tid & 31;
  const int wv   = tid >> 5;
  const int h = blockIdx.x % NHEADS;
  const int w = blockIdx.x / NHEADS;
  const int wi = w & 63;                 // mask window index
  const size_t wbase = (size_t)w * WTOK;

  for (int i = tid; i < 64 * LQ; i += 128) { qn[i] = (bf16_t)0.f; kn[i] = (bf16_t)0.f; }
  for (int i = tid; i < 32 * LV; i += 128) vt[i] = (bf16_t)0.f;
  for (int i = tid; i < 64 * LT; i += 128) at[i] = (bf16_t)0.f;
  for (int i = tid; i < 64 * 65; i += 128) lg[i] = -1e30f;
  __syncthreads();

  // q/k normalize (threads 0..63), v transpose (threads 64..127); vectorized 64B rows
  if (tid < 64) {
    const int t = tid;
    if (t < WTOK) {
      union { uint4 q4[4]; bf16_t hv[32]; } uq, uk;
      const uint4* qp = (const uint4*)(qkv + (wbase + t) * 1152 + h * HD);
      const uint4* kp = (const uint4*)(qkv + (wbase + t) * 1152 + 384 + h * HD);
#pragma unroll
      for (int c = 0; c < 4; ++c) { uq.q4[c] = qp[c]; uk.q4[c] = kp[c]; }
      float qs = 0.f, ks = 0.f;
#pragma unroll
      for (int d = 0; d < HD; ++d) {
        const float qv = (float)uq.hv[d], kv = (float)uk.hv[d];
        qs += qv * qv; ks += kv * kv;
      }
      const float qi = 1.f / fmaxf(sqrtf(qs), 1e-12f);
      const float ki = 1.f / fmaxf(sqrtf(ks), 1e-12f);
#pragma unroll
      for (int d = 0; d < HD; ++d) {
        qn[t * LQ + d] = (bf16_t)((float)uq.hv[d] * qi);
        kn[t * LQ + d] = (bf16_t)((float)uk.hv[d] * ki);
      }
    }
  } else {
    const int t = tid - 64;
    if (t < WTOK) {
      union { uint4 q4[4]; bf16_t hv[32]; } uv;
      const uint4* vp = (const uint4*)(qkv + (wbase + t) * 1152 + 768 + h * HD);
#pragma unroll
      for (int c = 0; c < 4; ++c) uv.q4[c] = vp[c];
#pragma unroll
      for (int d = 0; d < HD; ++d) vt[d * LV + t] = uv.hv[d];
    }
  }
  // pre-stage combined additive bias: rpb[h] + mask[wi] (valid 49x49 region)
  {
    const float* rp = rpb + (size_t)h * 2401;
    const float* mp = mask + (size_t)wi * 2401;
    for (int i = tid; i < 2401; i += 128) {
      const int m = i / WTOK, n = i % WTOK;
      lg[m * 65 + n] = rp[i] + mp[i];
    }
  }
  __syncthreads();

  const float scale = __expf(fminf(lscale[h], 4.6051701859880914f)); // ln(100)
  const v8f vzero = {0.f, 0.f, 0.f, 0.f, 0.f, 0.f, 0.f, 0.f};

  // QK^T: wave wv owns row tile [16*wv, 16*wv+16); epilogue is branch-free
  {
    Frag af;
    load_frag(&qn[(wv * 16) * LQ], LQ, lane, af);
#pragma unroll
    for (int nt = 0; nt < 4; ++nt) {
      Frag bfr;
      load_frag(&kn[(nt * 16) * LQ], LQ, lane, bfr);
      v8f c = vzero;
      c = __builtin_amdgcn_wmma_f32_16x16x32_bf16(false, af.v, false, bfr.v,
                                                  (short)0, c, false, false);
      const int n = nt * 16 + (lane & 15);
      const int mbase = wv * 16 + ((lane & 16) ? 8 : 0);
#pragma unroll
      for (int e = 0; e < 8; ++e)
        lg[(mbase + e) * 65 + n] += c[e] * scale;   // pads stay ~ -1e30 (acc==0)
    }
  }
  __syncthreads();

  // softmax over rows 0..48 (cols >=49 stay 0 in `at` from zero-init)
  if (tid < WTOK) {
    float mx = -1e30f;
    for (int n = 0; n < WTOK; ++n) mx = fmaxf(mx, lg[tid * 65 + n]);
    float s = 0.f;
    for (int n = 0; n < WTOK; ++n) s += __expf(lg[tid * 65 + n] - mx);
    const float inv = 1.f / s;
    for (int n = 0; n < WTOK; ++n)
      at[tid * LT + n] = (bf16_t)(__expf(lg[tid * 65 + n] - mx) * inv);
  }
  __syncthreads();

  // O = attn(64x64) @ v(64x32): 2 col tiles, K = 2x32
#pragma unroll
  for (int nt = 0; nt < 2; ++nt) {
    v8f c = vzero;
#pragma unroll
    for (int kk = 0; kk < 2; ++kk) {
      Frag af, bfr;
      load_frag(&at[(wv * 16) * LT + kk * 32], LT, lane, af);
      load_frag(&vt[(nt * 16) * LV + kk * 32], LV, lane, bfr);
      c = __builtin_amdgcn_wmma_f32_16x16x32_bf16(false, af.v, false, bfr.v,
                                                  (short)0, c, false, false);
    }
    const int n = nt * 16 + (lane & 15);
    const int mbase = wv * 16 + ((lane & 16) ? 8 : 0);
#pragma unroll
    for (int e = 0; e < 8; ++e) {
      const int m = mbase + e;
      if (m < WTOK)
        o[(wbase + m) * CDIM + h * HD + n] = (bf16_t)c[e];
    }
  }
}

// ---------------------------------------------------------------------------
// residual1: d_out = x + roll(window_reverse(proj_out))   (float4 elementwise)
// ---------------------------------------------------------------------------
__global__ __launch_bounds__(256)
void residual1_kernel(const float* __restrict__ x, const float* __restrict__ pout,
                      float* __restrict__ out) {
  const int idx = blockIdx.x * blockDim.x + threadIdx.x;  // over TOK*384/4
  const int t = idx / 96;                                 // token
  const int c4 = idx % 96;                                // float4 column
  const int b = t / 3136;
  const int pix = t % 3136;
  const int hh = pix / 56, ww = pix % 56;
  const int hs = (hh + 53) % 56;   // -3 mod 56
  const int wsft = (ww + 53) % 56;
  const int wnd = b * 64 + (hs / 7) * 8 + (wsft / 7);
  const int n = (hs % 7) * 7 + (wsft % 7);
  const size_t prow = ((size_t)wnd * WTOK + n) * CDIM + c4 * 4;
  const float4 xv = ((const float4*)x)[idx];
  const float4 pv = *(const float4*)(pout + prow);
  float4 r;
  r.x = xv.x + pv.x; r.y = xv.y + pv.y; r.z = xv.z + pv.z; r.w = xv.w + pv.w;
  ((float4*)out)[idx] = r;
}

// ---------------------------------------------------------------------------
// launch
// ---------------------------------------------------------------------------
extern "C" void kernel_launch(void* const* d_in, const int* in_sizes, int n_in,
                              void* d_out, int out_size, void* d_ws, size_t ws_size,
                              hipStream_t stream) {
  (void)in_sizes; (void)n_in; (void)out_size; (void)ws_size;
  const float* x        = (const float*)d_in[0];
  const float* amask    = (const float*)d_in[1];
  const float* n1g      = (const float*)d_in[2];
  const float* n1b      = (const float*)d_in[3];
  const float* qkv_w    = (const float*)d_in[4];
  const float* qkv_bias = (const float*)d_in[5];
  const float* proj_w   = (const float*)d_in[6];
  const float* proj_b   = (const float*)d_in[7];
  const float* cpb_w1   = (const float*)d_in[8];
  const float* cpb_b1   = (const float*)d_in[9];
  const float* cpb_w2   = (const float*)d_in[10];
  const float* lscale   = (const float*)d_in[11];
  const float* n2g      = (const float*)d_in[12];
  const float* n2b      = (const float*)d_in[13];
  const float* mlp_w1   = (const float*)d_in[14];
  const float* mlp_b1   = (const float*)d_in[15];
  const float* mlp_w2   = (const float*)d_in[16];
  const float* mlp_b2   = (const float*)d_in[17];
  float* out = (float*)d_out;

  char* wsb = (char*)d_ws;
  size_t off = 0;
  auto alloc = [&](size_t bytes) {
    void* p = wsb + off;
    off = (off + bytes + 255) & ~(size_t)255;
    return p;
  };
  bf16_t* wqkv  = (bf16_t*)alloc((size_t)1152 * 384 * 2);
  bf16_t* wproj = (bf16_t*)alloc((size_t)384 * 384 * 2);
  bf16_t* wmlp1 = (bf16_t*)alloc((size_t)768 * 384 * 2);
  bf16_t* wmlp2 = (bf16_t*)alloc((size_t)384 * 768 * 2);
  float*  tab   = (float*)alloc((size_t)169 * 12 * 4);
  float*  rpb   = (float*)alloc((size_t)12 * 49 * 49 * 4);
  bf16_t* xw    = (bf16_t*)alloc((size_t)TOK * CDIM * 2);     // LN1 out; reused for LN2 out
  bf16_t* qkvb  = (bf16_t*)alloc((size_t)TOK * 1152 * 2);     // reused as MLP hidden
  bf16_t* obuf  = (bf16_t*)alloc((size_t)TOK * CDIM * 2);
  float*  pbuf  = (float*)alloc((size_t)TOK * CDIM * 4);
  bf16_t* hidden = qkvb;

  // weights -> bf16
  f2bf_kernel<<<(1152 * 384 + 255) / 256, 256, 0, stream>>>(qkv_w, wqkv, 1152 * 384);
  f2bf_kernel<<<(384 * 384 + 255) / 256, 256, 0, stream>>>(proj_w, wproj, 384 * 384);
  f2bf_kernel<<<(768 * 384 + 255) / 256, 256, 0, stream>>>(mlp_w1, wmlp1, 768 * 384);
  f2bf_kernel<<<(384 * 768 + 255) / 256, 256, 0, stream>>>(mlp_w2, wmlp2, 384 * 768);

  // continuous position bias
  cpb1_kernel<<<169, 128, 0, stream>>>(cpb_w1, cpb_b1, cpb_w2, tab);
  cpb2_kernel<<<(12 * 49 * 49 + 255) / 256, 256, 0, stream>>>(tab, rpb);

  // LN1 + shift + window partition -> bf16
  ln_kernel<true><<<TOK, 128, 0, stream>>>(x, n1g, n1b, xw);

  // QKV GEMM: (TOK,384) @ (384,1152) -> bf16
  gemm_bf16_wmma<0><<<dim3(1152 / 128, TOK / 128), 256, 0, stream>>>(
      xw, wqkv, qkv_bias, nullptr, (void*)qkvb, TOK, 1152, 384);

  // attention per (window, head)
  attn_kernel<<<NWIN * NHEADS, 128, 0, stream>>>(qkvb, rpb, amask, lscale, obuf);

  // proj: (TOK,384) @ (384,384) -> fp32
  gemm_bf16_wmma<2><<<dim3(384 / 128, TOK / 128), 256, 0, stream>>>(
      obuf, wproj, proj_b, nullptr, (void*)pbuf, TOK, 384, 384);

  // residual 1 (window reverse + roll back)
  residual1_kernel<<<(TOK * CDIM / 4) / 256, 256, 0, stream>>>(x, pbuf, out);

  // LN2 -> bf16
  ln_kernel<false><<<TOK, 128, 0, stream>>>(out, n2g, n2b, xw);

  // MLP1 + GELU: (TOK,384) @ (384,768) -> bf16
  gemm_bf16_wmma<1><<<dim3(768 / 128, TOK / 128), 256, 0, stream>>>(
      xw, wmlp1, mlp_b1, nullptr, (void*)hidden, TOK, 768, 384);

  // MLP2 + residual: (TOK,768) @ (768,384) + d_out -> d_out
  gemm_bf16_wmma<3><<<dim3(384 / 128, TOK / 128), 256, 0, stream>>>(
      hidden, wmlp2, mlp_b2, out, (void*)out, TOK, 384, 768);
}
